// AttentionLayer_63007170233037
// MI455X (gfx1250) — matrix operand order repaired and checked by reading, not compiled, not run on hardware
//
#include <hip/hip_runtime.h>

// ---------------------------------------------------------------------------
// Fused conv-QKV + LN + attention + projection for MI455X (gfx1250, wave32).
// All matmuls via v_wmma_f32_16x16x32_bf16; x tile staged via the Tensor
// Data Mover (TENSOR_LOAD_TO_LDS) when the builtin is available.
// ---------------------------------------------------------------------------

typedef __attribute__((ext_vector_type(16))) __bf16 v16bf;
typedef __attribute__((ext_vector_type(8)))  float  v8f;
typedef __attribute__((ext_vector_type(4)))  unsigned int v4u;
typedef __attribute__((ext_vector_type(8)))  int    v8i;
typedef __attribute__((ext_vector_type(4)))  int    v4i;

union Frag16 { v16bf v; uint4 u[2]; };
static_assert(sizeof(Frag16) == 32, "frag size");

static __device__ __forceinline__ v8f wmma_bf16(v16bf a, v16bf b, v8f c) {
  return __builtin_amdgcn_wmma_f32_16x16x32_bf16(false, a, false, b, (short)0, c,
                                                 false, false);
}

#if defined(__has_builtin)
#if __has_builtin(__builtin_amdgcn_tensor_load_to_lds) && \
    __has_builtin(__builtin_amdgcn_s_wait_tensorcnt)
#define HAVE_TDM 1
#endif
#endif
#ifndef HAVE_TDM
#define HAVE_TDM 0
#endif

#define TPB 256

// LDS layout (bytes)
constexpr int OFF_BIAS = 0;                       // 8*20*20*4      = 12800
constexpr int OFF_XS   = 12800;                   // 4*22*256*2     = 45056
constexpr int OFF_CONV = 57856;                   // 80*256*4       = 81920 (aliased: x f32 stage, ao bf16)
constexpr int OFF_Q    = 139776;                  // 80*256*2       = 40960
constexpr int OFF_K    = 180736;                  // 80*256*2       = 40960
constexpr int OFF_VT   = 221696;                  // 4*256*32*2     = 65536
constexpr int OFF_SC   = 287232;                  // 8*32*32*2      = 16384
constexpr int SMEM_BYTES = 303616;                // <= 320KB per WGP

// ws layout: conv weight fragments (bf16) then wo (bf16)
constexpr int WB_ELEMS   = 3 * 768 * 256;         // 589824 bf16
constexpr int WOB_OFFSET = WB_ELEMS * 2;          // bytes

// ---------------------------------------------------------------------------
// Weight pre-swizzle: conv weights -> the WMMA B-fragment lane order this kernel uses (same k order as A).
// B (32x16, bf16) as packed here: lane L (col = L&15) holds 16 contiguous K values,
// K = (L>>4)*16 + e.  Fragment storage: tile(t=kstep,u=otile) base
// (t*16+u)*512 elems, lane*16 + e.
// ---------------------------------------------------------------------------
__global__ void convert_weights(const float* __restrict__ wq,
                                const float* __restrict__ wk,
                                const float* __restrict__ wv,
                                const float* __restrict__ wo,
                                __bf16* __restrict__ wb,
                                __bf16* __restrict__ wob) {
  int idx = blockIdx.x * TPB + threadIdx.x;
  if (idx < WB_ELEMS) {
    int c    = idx / 196608;
    int rem  = idx - c * 196608;
    int tile = rem >> 9;
    int t = tile >> 4, u = tile & 15;
    int lane = (rem >> 4) & 31;
    int e    = rem & 15;
    int K  = ((lane >> 4) << 4) + e;      // half*16 + e (B: contiguous 16 K)
    int kk = t * 32 + K;                  // 0..767 : kk = k*256 + i
    int k  = kk >> 8, i = kk & 255;
    int o  = u * 16 + (lane & 15);
    const float* w = (c == 0) ? wq : ((c == 1) ? wk : wv);
    wb[idx] = (__bf16)w[(o * 256 + i) * 3 + k];
  }
  if (idx < 256 * 256) wob[idx] = (__bf16)wo[idx];
}

// ---------------------------------------------------------------------------
// Fused kernel: one WG = 4 batch elements (80 rows).
// ---------------------------------------------------------------------------
__global__ __launch_bounds__(TPB, 1)
void fused_attn(const float* __restrict__ x,
                const float* __restrict__ gq, const float* __restrict__ bq,
                const float* __restrict__ gk, const float* __restrict__ bk,
                const float* __restrict__ gvn, const float* __restrict__ bvn,
                const float* __restrict__ rel, const float* __restrict__ gbias,
                const float* __restrict__ alphaP,
                const float* __restrict__ bo,
                const __bf16* __restrict__ wb,
                const __bf16* __restrict__ wob,
                float* __restrict__ out) {
  extern __shared__ char smem[];
  float*  biasS = (float*) (smem + OFF_BIAS);
  __bf16* xs    = (__bf16*)(smem + OFF_XS);    // [4][22][256] rows n=-1..20
  float*  convS = (float*) (smem + OFF_CONV);  // [80][256] f32 scratch
  __bf16* aoS   = (__bf16*)(smem + OFF_CONV);  // alias: attn out [80][256] bf16
  __bf16* qS    = (__bf16*)(smem + OFF_Q);     // [80][256]
  __bf16* kS    = (__bf16*)(smem + OFF_K);     // [80][256]
  __bf16* vT    = (__bf16*)(smem + OFF_VT);    // [4][256][32]  (d-major, m padded)
  __bf16* scS   = (__bf16*)(smem + OFF_SC);    // [8 waves][32][32]

  const int tid  = threadIdx.x;
  const int w    = tid >> 5;
  const int lane = tid & 31;
  const int half = lane >> 4;
  const int l16  = lane & 15;
  const int blk  = blockIdx.x;
  const long long xbase = (long long)blk * (4 * 20 * 256);

  // ---- combined score bias table: rel + alpha*gbias ----
  {
    float alpha = alphaP[0];
    for (int i = tid; i < 8 * 20 * 20; i += TPB) {
      int h = i / 400;
      int r = (i / 20) % 20;
      int c = i % 20;
      biasS[i] = rel[(r - c + 19) * 8 + h] + alpha * gbias[i];
    }
  }
  // ---- zero vT (covers the m=20..31 padding) ----
  {
    uint* vz = (uint*)vT;
    for (int i = tid; i < 4 * 256 * 32 / 2; i += TPB) vz[i] = 0u;
  }

#if HAVE_TDM
  // ---- stage x tile (80 rows x 256 f32 = 80KB) via Tensor Data Mover ----
  if (w == 0) {
    unsigned long long ga = (unsigned long long)(const void*)(x + xbase);
    v4u g0;
    g0[0] = 1u;                                   // count=1, user descriptor
    g0[1] = (unsigned)OFF_CONV;                   // lds_addr (bytes)
    g0[2] = (unsigned)ga;                         // global_addr[31:0]
    g0[3] = ((unsigned)(ga >> 32) & 0x01FFFFFFu)  // global_addr[56:32]
            | 0x80000000u;                        // type = 2 ("image")
    v8i g1;
    g1[0] = 2 << 16;                              // data_size = 4B, no mask
    g1[1] = (int)(256u << 16);                    // tensor_dim0 = 256
    g1[2] = (int)(80u << 16);                     // tensor_dim1 = 80
    g1[3] = (int)(256u << 16);                    // tile_dim0 = 256
    g1[4] = 80;                                   // tile_dim1 = 80, tile_dim2=0
    g1[5] = 256;                                  // tensor_dim0_stride = 256
    g1[6] = 0;
    g1[7] = 0;
    v4i gz = {0, 0, 0, 0};
#if __clang_major__ >= 23
    v8i gz8 = {0, 0, 0, 0, 0, 0, 0, 0};
    __builtin_amdgcn_tensor_load_to_lds(g0, g1, gz, gz, gz8, 0);
#else
    __builtin_amdgcn_tensor_load_to_lds(g0, g1, gz, gz, 0);
#endif
    __builtin_amdgcn_s_wait_tensorcnt(0);
  }
  __syncthreads();
  // convert f32 tile -> bf16 xs with zero halo rows
  for (int i = tid; i < 4 * 22 * 256; i += TPB) {
    int b = i / (22 * 256);
    int r = (i >> 8) % 22;
    int d = i & 255;
    int n = r - 1;
    float v = (n >= 0 && n < 20) ? convS[((b * 20 + n) << 8) + d] : 0.f;
    xs[i] = (__bf16)v;
  }
#else
  // ---- fallback: scalar staging of x into LDS as bf16 with halo ----
  for (int i = tid; i < 4 * 22 * 256; i += TPB) {
    int b = i / (22 * 256);
    int r = (i >> 8) % 22;
    int d = i & 255;
    int n = r - 1;
    float v = 0.f;
    if (n >= 0 && n < 20) v = x[xbase + ((b * 20 + n) << 8) + d];
    xs[i] = (__bf16)v;
  }
#endif
  __syncthreads();

  // =========================================================================
  // conv1d(K=3) as 80x768x256 GEMM, then LayerNorm + residual, for q/k/v
  // =========================================================================
  for (int c = 0; c < 3; ++c) {
    const __bf16* W = wb + c * 196608;
    #pragma unroll 1
    for (int t = 0; t < 10; ++t) {            // 5 Mtiles x 16 Ntiles / 8 waves
      int tix = w * 10 + t;
      int mt = tix >> 4, ot = tix & 15;
      int arow = mt * 16 + l16;               // 0..79, always valid
      const __bf16* abase = xs + ((arow / 20) * 22 + (arow % 20)) * 256;
      v8f acc = {};
      #pragma unroll
      for (int ks = 0; ks < 24; ++ks) {
        Frag16 A, B;
        int kk1 = ks * 32 + half * 16;        // A: contiguous 16 K per lane
        A.u[0] = *(const uint4*)(abase + kk1);
        A.u[1] = *(const uint4*)(abase + kk1 + 8);
        const __bf16* wptr = W + (((ks * 16 + ot) << 9) + (lane << 4));
        __builtin_prefetch(wptr + 8192, 0, 1);
        B.u[0] = ((const uint4*)wptr)[0];
        B.u[1] = ((const uint4*)wptr)[1];
        acc = wmma_bf16(A.v, B.v, acc);
      }
      int col = ot * 16 + l16;
      int r0  = mt * 16 + 8 * half;
      #pragma unroll
      for (int j = 0; j < 8; ++j) convS[(r0 + j) * 256 + col] = acc[j];
    }
    __syncthreads();

    // LayerNorm over D=256 per row + residual, write bf16 (v transposed)
    const float* g  = (c == 0) ? gq : ((c == 1) ? gk : gvn);
    const float* be = (c == 0) ? bq : ((c == 1) ? bk : bvn);
    #pragma unroll 1
    for (int rr = 0; rr < 10; ++rr) {
      int r = w * 10 + rr;
      float vals[8], s = 0.f, s2 = 0.f;
      #pragma unroll
      for (int e = 0; e < 8; ++e) {
        float v = convS[r * 256 + e * 32 + lane];
        vals[e] = v; s += v; s2 += v * v;
      }
      #pragma unroll
      for (int off = 16; off; off >>= 1) {
        s  += __shfl_xor(s,  off);
        s2 += __shfl_xor(s2, off);
      }
      float mean = s * (1.f / 256.f);
      float inv  = rsqrtf(s2 * (1.f / 256.f) - mean * mean + 1e-5f);
      int b = r / 20, n = r % 20;
      #pragma unroll
      for (int e = 0; e < 8; ++e) {
        int d = e * 32 + lane;
        float y = (vals[e] - mean) * inv * g[d] + be[d]
                + (float)xs[(b * 22 + n + 1) * 256 + d];
        if (c == 2)      vT[(((b << 8) + d) << 5) + n] = (__bf16)y;
        else if (c == 0) qS[r * 256 + d] = (__bf16)y;
        else             kS[r * 256 + d] = (__bf16)y;
      }
    }
    __syncthreads();
  }

  // =========================================================================
  // attention: per (batch,head), 20x20 scores padded to 32x32 (2x2 WMMA tiles)
  // =========================================================================
  const uint4 zz = {0u, 0u, 0u, 0u};
  #pragma unroll 1
  for (int p = w; p < 32; p += 8) {
    int b = p >> 3, h = p & 7;

    Frag16 qf[2], kf[2];
    #pragma unroll
    for (int mt = 0; mt < 2; ++mt) {          // A operand: q rows
      int qrow = mt * 16 + l16;
      bool valid = qrow < 20;
      const __bf16* qp = qS + ((b * 20 + (valid ? qrow : 0)) << 8)
                            + h * 32 + half * 16;
      qf[mt].u[0] = valid ? *(const uint4*)qp        : zz;
      qf[mt].u[1] = valid ? *(const uint4*)(qp + 8) : zz;
    }
    #pragma unroll
    for (int nt = 0; nt < 2; ++nt) {          // B operand: k^T columns
      int kcol = nt * 16 + l16;
      bool valid = kcol < 20;
      const __bf16* kp = kS + ((b * 20 + (valid ? kcol : 0)) << 8)
                            + h * 32 + half * 16;
      kf[nt].u[0] = valid ? *(const uint4*)kp       : zz;
      kf[nt].u[1] = valid ? *(const uint4*)(kp + 8) : zz;
    }
    v8f sacc[2][2];
    #pragma unroll
    for (int mt = 0; mt < 2; ++mt)
      #pragma unroll
      for (int nt = 0; nt < 2; ++nt) {
        v8f zc = {};
        sacc[mt][nt] = wmma_bf16(qf[mt].v, kf[nt].v, zc);
      }

    // softmax over 20 cols (masked), fused biases; result -> LDS bf16
    const float scale = 0.17677669529663687f;   // 1/sqrt(32)
    __bf16* sc = scS + (w << 10);
    #pragma unroll
    for (int mt = 0; mt < 2; ++mt) {
      #pragma unroll
      for (int j = 0; j < 8; ++j) {
        int row = mt * 16 + j + 8 * half;
        int c0 = l16, c1 = l16 + 16;
        bool rv = row < 20;
        int  rc = rv ? row : 0;
        float t0 = rv ? sacc[mt][0][j] * scale + biasS[h * 400 + rc * 20 + c0]
                      : -1e30f;
        bool cv1 = c1 < 20;
        float t1 = (rv && cv1)
                 ? sacc[mt][1][j] * scale + biasS[h * 400 + rc * 20 + (cv1 ? c1 : 0)]
                 : -1e30f;
        float m = fmaxf(t0, t1);
        m = fmaxf(m, __shfl_xor(m, 1));
        m = fmaxf(m, __shfl_xor(m, 2));
        m = fmaxf(m, __shfl_xor(m, 4));
        m = fmaxf(m, __shfl_xor(m, 8));
        float e0 = __expf(t0 - m), e1 = __expf(t1 - m);
        float su = e0 + e1;
        su += __shfl_xor(su, 1);
        su += __shfl_xor(su, 2);
        su += __shfl_xor(su, 4);
        su += __shfl_xor(su, 8);
        float inv = rv ? (1.0f / su) : 0.0f;
        sc[row * 32 + c0] = (__bf16)(e0 * inv);
        sc[row * 32 + c1] = (__bf16)(e1 * inv);
      }
    }

    // out = attn @ v   (K = 32 padded positions, zeros beyond 20)
    Frag16 af[2], vf[2];
    #pragma unroll
    for (int mt = 0; mt < 2; ++mt) {
      const __bf16* ap = sc + (mt * 16 + l16) * 32 + half * 16;
      af[mt].u[0] = *(const uint4*)ap;
      af[mt].u[1] = *(const uint4*)(ap + 8);
    }
    #pragma unroll
    for (int nt = 0; nt < 2; ++nt) {
      const __bf16* vp = vT + (((b << 8) + h * 32 + nt * 16 + l16) << 5)
                            + half * 16;
      vf[nt].u[0] = *(const uint4*)vp;
      vf[nt].u[1] = *(const uint4*)(vp + 8);
    }
    #pragma unroll
    for (int mt = 0; mt < 2; ++mt)
      #pragma unroll
      for (int nt = 0; nt < 2; ++nt) {
        v8f zc = {};
        v8f o = wmma_bf16(af[mt].v, vf[nt].v, zc);
        #pragma unroll
        for (int j = 0; j < 8; ++j) {
          int row = mt * 16 + j + 8 * half;
          if (row < 20)
            aoS[((b * 20 + row) << 8) + h * 32 + nt * 16 + l16] = (__bf16)o[j];
        }
      }
  }
  __syncthreads();

  // =========================================================================
  // output projection: [80,256] x wo^T (256x256) + bo
  // =========================================================================
  #pragma unroll 1
  for (int t = 0; t < 10; ++t) {
    int tix = w * 10 + t;
    int mt = tix >> 4, ot = tix & 15;
    const __bf16* ab = aoS + ((mt * 16 + l16) << 8);
    v8f acc = {};
    #pragma unroll
    for (int ks = 0; ks < 8; ++ks) {
      Frag16 A, B;
      int kb = ks * 32 + half * 16;
      A.u[0] = *(const uint4*)(ab + kb);
      A.u[1] = *(const uint4*)(ab + kb + 8);
      const __bf16* wp = wob + ((ot * 16 + l16) << 8) + ks * 32 + half * 16;
      B.u[0] = *(const uint4*)wp;
      B.u[1] = *(const uint4*)(wp + 8);
      acc = wmma_bf16(A.v, B.v, acc);
    }
    int col = ot * 16 + l16;
    float bias = bo[col];
    const long long obase = (long long)blk * (80 * 256);
    #pragma unroll
    for (int j = 0; j < 8; ++j) {
      int r = mt * 16 + j + 8 * half;
      out[obase + r * 256 + col] = acc[j] + bias;
    }
  }
}

// ---------------------------------------------------------------------------
extern "C" void kernel_launch(void* const* d_in, const int* in_sizes, int n_in,
                              void* d_out, int out_size, void* d_ws, size_t ws_size,
                              hipStream_t stream) {
  (void)in_sizes; (void)n_in; (void)out_size; (void)ws_size;
  const float* x   = (const float*)d_in[0];
  const float* wq  = (const float*)d_in[1];
  const float* wk  = (const float*)d_in[2];
  const float* wv  = (const float*)d_in[3];
  const float* gq  = (const float*)d_in[4];
  const float* bq  = (const float*)d_in[5];
  const float* gk  = (const float*)d_in[6];
  const float* bk  = (const float*)d_in[7];
  const float* gv  = (const float*)d_in[8];
  const float* bv  = (const float*)d_in[9];
  const float* rel = (const float*)d_in[10];
  const float* gb  = (const float*)d_in[11];
  const float* al  = (const float*)d_in[12];
  const float* wo  = (const float*)d_in[13];
  const float* bo  = (const float*)d_in[14];

  __bf16* wb  = (__bf16*)d_ws;
  __bf16* wob = (__bf16*)((char*)d_ws + WOB_OFFSET);

  hipFuncSetAttribute(reinterpret_cast<const void*>(fused_attn),
                      hipFuncAttributeMaxDynamicSharedMemorySize, SMEM_BYTES);

  convert_weights<<<(WB_ELEMS + TPB - 1) / TPB, TPB, 0, stream>>>(
      wq, wk, wv, wo, wb, wob);

  fused_attn<<<8192 / 4, TPB, SMEM_BYTES, stream>>>(
      x, gq, bq, gk, bk, gv, bv, rel, gb, al, bo, wb, wob, (float*)d_out);
}
